// LinearAttention_1065151889826
// MI455X (gfx1250) — matrix operand order repaired and checked by reading, not compiled, not run on hardware
//
#include <hip/hip_runtime.h>

// ---------------------------------------------------------------------------
// Linear attention, MI455X (gfx1250, wave32, WMMA 16x16x32 bf16).
//
// reference:  q = softmax((Wq query)/T, axis=spatial)
//             k = softmax((Wk key)*mask/T, axis=channel)
//             v = (Wv value)*mask
//             out = q^T-contract( k v^T )        (never form 4096x32768 attn)
//
// mask keeps only the last 5 frames: columns s in [12288, 32768) -> 20480 cols.
// masked softk columns are uniform but multiply v==0 -> skip them entirely.
// ---------------------------------------------------------------------------

typedef unsigned short ushort_t;
typedef unsigned int   uint_t;
typedef __bf16 bf16_t;
typedef __attribute__((ext_vector_type(16))) __bf16 v16bf;
typedef __attribute__((ext_vector_type(8)))  float  v8f;

union FragAB { v16bf v; uint_t u[8]; ushort_t s[16]; };

#define C_DIM   256
#define HW_DIM  4096
#define S_DIM   32768
#define S0_KEEP 12288
#define SK_KEEP 20480           // kept columns (last 5 frames)
#define NSPLIT  40              // split-K factor for the KV GEMM
#define INV_T   10.0f           // 1 / TEMP_SCALE

__device__ __forceinline__ ushort_t f2bf(float x) {
    // round-to-nearest-even fp32 -> bf16
    uint_t u = __float_as_uint(x);
    u += 0x7FFFu + ((u >> 16) & 1u);
    return (ushort_t)(u >> 16);
}
__device__ __forceinline__ ushort_t to_bf16(float x)    { return f2bf(x); }
__device__ __forceinline__ ushort_t to_bf16(ushort_t x) { return x; }

__device__ __forceinline__ void store_out(float* p, float v)    { *p = v; }
__device__ __forceinline__ void store_out(ushort_t* p, float v) { *p = f2bf(v); }

// ---------------------------------------------------------------------------
// Generic 256xN WMMA GEMM:  D(256 x ncols) = A(256 x Ktot) * B(Ktot x ncols)
//   A[m][k], row stride lda (k contiguous)            AT in {float, bf16 bits}
//   B: BTRANS ? B[n][k] (ldb row stride) : B[k][n]    BT in {float, bf16 bits}
//   blockIdx.x -> 64-column slab; blockIdx.y -> split-K slice of kSplit.
// Per block: 8 waves, wave w owns M-blocks {2w,2w+1} x 4 N-blocks
//   -> 8 v8f accumulators/wave, v_wmma_f32_16x16x32_bf16.
// ---------------------------------------------------------------------------
template <typename AT, typename BT, typename OT, bool BTRANS>
__global__ __launch_bounds__(256) void gemm_wmma(
    const AT* __restrict__ A, long lda,
    const BT* __restrict__ B, long ldb,
    OT* __restrict__ D, long ldd,
    int kSplit, long dSplitStride)
{
    __shared__ __attribute__((aligned(16))) ushort_t aS[256][34]; // 256 x 32 bf16 (+pad)
    __shared__ __attribute__((aligned(16))) ushort_t bS[64][34];  //  64 x 32 bf16 (+pad)

    const int tid  = threadIdx.x;
    const int wave = tid >> 5;
    const int lane = tid & 31;
    const int h    = lane >> 4;   // lane half
    const int ln   = lane & 15;

    const long col0 = (long)blockIdx.x * 64;
    const long koff = (long)blockIdx.y * (long)kSplit;
    A += koff;                                  // k contiguous in A
    B += BTRANS ? koff : koff * ldb;
    D += (long)blockIdx.y * dSplitStride;

    v8f acc[2][4];
#pragma unroll
    for (int mb = 0; mb < 2; ++mb)
#pragma unroll
        for (int nb = 0; nb < 4; ++nb)
#pragma unroll
            for (int r = 0; r < 8; ++r) acc[mb][nb][r] = 0.0f;

    const int mb0 = wave * 2;

    for (int kc = 0; kc < kSplit; kc += 32) {
        // ---- stage A chunk (256 x 32) into LDS as bf16, coalesced ----
#pragma unroll
        for (int i = 0; i < 32; ++i) {
            int e = i * 256 + tid;
            int r = e >> 5, k = e & 31;
            aS[r][k] = to_bf16(A[(long)r * lda + kc + k]);
        }
        // ---- stage B chunk (32 x 64) into LDS as bf16 [n][k] ----
#pragma unroll
        for (int i = 0; i < 8; ++i) {
            int e = i * 256 + tid;
            if (BTRANS) {
                int n = e >> 5, k = e & 31;   // consecutive tid -> consecutive k
                bS[n][k] = to_bf16(B[(col0 + n) * ldb + kc + k]);
            } else {
                int k = e >> 6, n = e & 63;   // consecutive tid -> consecutive n
                bS[n][k] = to_bf16(B[(long)(kc + k) * ldb + col0 + n]);
            }
        }
        __syncthreads();

        // ---- B fragments (32x16), ISA layout: VGPR j pair K = (j>>2)*16 + 8h + (2j&7) of col n
        FragAB bf[4];
#pragma unroll
        for (int nb = 0; nb < 4; ++nb) {
            int n = nb * 16 + ln;
#pragma unroll
            for (int j = 0; j < 8; ++j) {
                int K = ((j >> 2) * 16) + h * 8 + ((2 * j) & 7);
                bf[nb].u[j] = *(const uint_t*)&bS[n][K];
            }
        }
        // ---- A fragments (16x32), ISA layout: VGPR j pair K = (j>>2)*16 + 8h + (2j&7)
#pragma unroll
        for (int mb = 0; mb < 2; ++mb) {
            FragAB af;
            int row = (mb0 + mb) * 16 + ln;
#pragma unroll
            for (int j = 0; j < 8; ++j) {
                int K = ((j >> 2) * 16) + h * 8 + ((2 * j) & 7);
                af.u[j] = *(const uint_t*)&aS[row][K];
            }
#pragma unroll
            for (int nb = 0; nb < 4; ++nb) {
                acc[mb][nb] = __builtin_amdgcn_wmma_f32_16x16x32_bf16(
                    false, af.v, false, bf[nb].v, (short)0, acc[mb][nb],
                    false, false);
            }
        }
        __syncthreads();
    }

    // ---- store D: VGPR r -> M = 8h + r, N = ln (ISA C/D layout) ----
#pragma unroll
    for (int mb = 0; mb < 2; ++mb)
#pragma unroll
        for (int nb = 0; nb < 4; ++nb)
#pragma unroll
            for (int r = 0; r < 8; ++r) {
                long m = (long)(mb0 + mb) * 16 + h * 8 + r;
                long n = col0 + nb * 16 + ln;
                store_out(&D[m * ldd + n], acc[mb][nb][r]);
            }
}

// ---------------------------------------------------------------------------
// Channel softmax (axis = C) per kept column, temp 0.1, output bf16.
// Block handles 64 columns; 4 partial threads per column.
// ---------------------------------------------------------------------------
__global__ __launch_bounds__(256) void softmax_channel(
    const float* __restrict__ kp, ushort_t* __restrict__ softk, int ld)
{
    __shared__ float red[4][64];
    __shared__ float cmax[64];
    __shared__ float cinv[64];
    const int tid  = threadIdx.x;
    const int col  = blockIdx.x * 64 + (tid & 63);
    const int part = tid >> 6;

    float m = -1e30f;
    for (int c = part * 64; c < part * 64 + 64; ++c)
        m = fmaxf(m, kp[(long)c * ld + col]);
    red[part][tid & 63] = m;
    __syncthreads();
    if (tid < 64)
        cmax[tid] = fmaxf(fmaxf(red[0][tid], red[1][tid]),
                          fmaxf(red[2][tid], red[3][tid]));
    __syncthreads();
    const float M = cmax[tid & 63];

    float s = 0.0f;
    for (int c = part * 64; c < part * 64 + 64; ++c)
        s += __expf((kp[(long)c * ld + col] - M) * INV_T);
    red[part][tid & 63] = s;
    __syncthreads();
    if (tid < 64)
        cinv[tid] = 1.0f / (red[0][tid] + red[1][tid] + red[2][tid] + red[3][tid]);
    __syncthreads();
    const float inv = cinv[tid & 63];

    for (int c = part * 64; c < part * 64 + 64; ++c)
        softk[(long)c * ld + col] =
            f2bf(__expf((kp[(long)c * ld + col] - M) * INV_T) * inv);
}

// ---------------------------------------------------------------------------
// Row softmax (axis = spatial, 4096 wide) per channel, temp 0.1, output bf16.
// One block per channel row.
// ---------------------------------------------------------------------------
__global__ __launch_bounds__(256) void softmax_row(
    const float* __restrict__ qp, ushort_t* __restrict__ softq, int n)
{
    __shared__ float red[256];
    const int tid = threadIdx.x;
    const long base = (long)blockIdx.x * n;

    float m = -1e30f;
    for (int i = tid; i < n; i += 256) m = fmaxf(m, qp[base + i]);
    red[tid] = m;
    __syncthreads();
    for (int s = 128; s > 0; s >>= 1) {
        if (tid < s) red[tid] = fmaxf(red[tid], red[tid + s]);
        __syncthreads();
    }
    const float M = red[0];
    __syncthreads();

    float s = 0.0f;
    for (int i = tid; i < n; i += 256) s += __expf((qp[base + i] - M) * INV_T);
    red[tid] = s;
    __syncthreads();
    for (int st = 128; st > 0; st >>= 1) {
        if (tid < st) red[tid] += red[tid + st];
        __syncthreads();
    }
    const float inv = 1.0f / red[0];

    for (int i = tid; i < n; i += 256)
        softq[base + i] = f2bf(__expf((qp[base + i] - M) * INV_T) * inv);
}

// ---------------------------------------------------------------------------
// Deterministic split-K reduction of KV partials; emit KV^T in bf16 so the
// final GEMM's A operand has its contraction index (c') contiguous.
// ---------------------------------------------------------------------------
__global__ __launch_bounds__(256) void reduce_kv(
    const float* __restrict__ part, ushort_t* __restrict__ KVt)
{
    const int c  = threadIdx.x;   // column of KV (row of KVt)
    const int cp = blockIdx.x;    // c'
    float s = 0.0f;
#pragma unroll 4
    for (int k = 0; k < NSPLIT; ++k)
        s += part[(long)k * (C_DIM * C_DIM) + (long)cp * C_DIM + c];
    KVt[(long)c * C_DIM + cp] = f2bf(s);
}

// ---------------------------------------------------------------------------
extern "C" void kernel_launch(void* const* d_in, const int* in_sizes, int n_in,
                              void* d_out, int out_size, void* d_ws, size_t ws_size,
                              hipStream_t stream)
{
    const float* query = (const float*)d_in[0];   // (256, 4096)
    const float* key   = (const float*)d_in[1];   // (256, 32768)
    const float* value = (const float*)d_in[2];   // (256, 32768)
    const float* Wq    = (const float*)d_in[3];   // (256, 256)
    const float* Wk    = (const float*)d_in[4];
    const float* Wv    = (const float*)d_in[5];
    float* out = (float*)d_out;                   // (256, 4096)

    char* ws = (char*)d_ws;
    // region 0 (20 MB): kp fp32, later reused for KV partials + qp + softq
    float*    kp    = (float*)(ws);                               // 256*20480 f32
    float*    part  = (float*)(ws);                               // 40*256*256 f32 (after K3)
    float*    qp    = (float*)(ws + 10485760);                    // 256*4096 f32
    ushort_t* softq = (ushort_t*)(ws + 10485760 + 4194304);       // 256*4096 bf16
    ushort_t* vp    = (ushort_t*)(ws + 20971520);                 // 256*20480 bf16
    ushort_t* softk = (ushort_t*)(ws + 31457280);                 // 256*20480 bf16
    ushort_t* KVt   = (ushort_t*)(ws + 41943040);                 // 256*256 bf16

    const dim3 blk(256);

    // 1) kp = Wk * key[:, 12288:]          (256x256 @ 256x20480)
    gemm_wmma<float, float, float, false><<<dim3(SK_KEEP / 64, 1), blk, 0, stream>>>(
        Wk, C_DIM, key + S0_KEEP, S_DIM, kp, SK_KEEP, C_DIM, 0);
    // 2) vp = Wv * value[:, 12288:]  -> bf16
    gemm_wmma<float, float, ushort_t, false><<<dim3(SK_KEEP / 64, 1), blk, 0, stream>>>(
        Wv, C_DIM, value + S0_KEEP, S_DIM, vp, SK_KEEP, C_DIM, 0);
    // 3) softk = softmax_channel(kp / T)
    softmax_channel<<<dim3(SK_KEEP / 64), blk, 0, stream>>>(kp, softk, SK_KEEP);
    // 4) KV partials: softk (256 x 20480) * vp^T (20480 x 256), split-K = 40
    gemm_wmma<ushort_t, ushort_t, float, true><<<dim3(4, NSPLIT), blk, 0, stream>>>(
        softk, SK_KEEP, vp, SK_KEEP, part, C_DIM, SK_KEEP / NSPLIT,
        (long)C_DIM * C_DIM);
    // 5) KVt = (sum_k partials)^T  -> bf16
    reduce_kv<<<dim3(C_DIM), blk, 0, stream>>>(part, KVt);
    // 6) qp = Wq * query
    gemm_wmma<float, float, float, false><<<dim3(HW_DIM / 64, 1), blk, 0, stream>>>(
        Wq, C_DIM, query, HW_DIM, qp, HW_DIM, C_DIM, 0);
    // 7) softq = softmax_row(qp / T)
    softmax_row<<<dim3(C_DIM), blk, 0, stream>>>(qp, softq, HW_DIM);
    // 8) out = KVt (256x256) * softq (256x4096)
    gemm_wmma<ushort_t, ushort_t, float, false><<<dim3(HW_DIM / 64, 1), blk, 0, stream>>>(
        KVt, C_DIM, softq, HW_DIM, out, HW_DIM, C_DIM, 0);
}
